// MemoryMolecular_27255862460912
// MI455X (gfx1250) — matrix operand order repaired and checked by reading, not compiled, not run on hardware
//
#include <hip/hip_runtime.h>
#include <hip/hip_bf16.h>
#include <stdint.h>

// ---------------------------------------------------------------------------
// B=2048, K=65536, F=512 fp32. Compute-bound (AI ~930 FLOP/B) =>
// bf16 WMMA GEMM with fused argmax/argmin, feature_queue tile staged in LDS
// via CDNA5 async-to-LDS loads (ASYNCcnt path), then row gather.
// ---------------------------------------------------------------------------
#define B_DIM 2048
#define K_DIM 65536
#define F_DIM 512
#define NT    8                  // 16-wide N tiles per wave -> 128 cols
#define KSTEP 32                 // K depth of v_wmma_f32_16x16x32_bf16
#define BLK_M 128                // 8 waves x 16 rows
#define BLK_N 128                // shared LDS tile of feature_queue rows
#define ROW_PAD 8                // +16B per LDS row: banks 4*row apart
#define ROW_ELT (F_DIM + ROW_PAD) // 520 uint16 per LDS row

typedef __attribute__((ext_vector_type(16))) __bf16 bf16x16;
typedef __attribute__((ext_vector_type(8)))  __bf16 bf16x8;
typedef __attribute__((ext_vector_type(8)))  float  f32x8;

// Monotonic unsigned key for float ordering (total order incl. negatives).
__device__ __forceinline__ uint32_t f32_ordkey(float f) {
    uint32_t u = __float_as_uint(f);
    return (u & 0x80000000u) ? ~u : (u | 0x80000000u);
}

__device__ __forceinline__ uint16_t f32_to_bf16_rne(float f) {
    uint32_t u = __float_as_uint(f);
    return (uint16_t)((u + 0x7FFFu + ((u >> 16) & 1u)) >> 16);
}

// ---------------------------------------------------------------------------
__global__ void mm_init_kernel(unsigned long long* __restrict__ pmax,
                               unsigned long long* __restrict__ pmin) {
    int i = blockIdx.x * blockDim.x + threadIdx.x;
    if (i < B_DIM) {
        pmax[i] = 0ull;
        pmin[i] = ~0ull;
    }
}

// ---------------------------------------------------------------------------
__global__ void mm_cvt4_kernel(const float4* __restrict__ src,
                               ushort4* __restrict__ dst, int n4) {
    int i = blockIdx.x * blockDim.x + threadIdx.x;
    if (i >= n4) return;
    float4 f = src[i];
    ushort4 o;
    o.x = f32_to_bf16_rne(f.x);
    o.y = f32_to_bf16_rne(f.y);
    o.z = f32_to_bf16_rne(f.z);
    o.w = f32_to_bf16_rne(f.w);
    dst[i] = o;
}

// ---------------------------------------------------------------------------
// Fused bf16 WMMA GEMM + per-row argmax/argmin.
//
// Block (256 thr, 8 waves) owns a 128(M) x 128(N) logits tile. The 128
// feature_queue rows (x full F = 130KB, padded) are staged ONCE into LDS
// with global_load_async_to_lds_b128 (coalesced 16B chunks, ASYNCcnt),
// then all 8 waves stream B fragments from LDS (ds_load_b128; padding
// makes the 16-row x 16B pattern hit each of the 64 banks exactly twice).
//
// Per wave: 16(M) x 128(N), 8 f32x8 accumulators, 8 WMMA per 32-deep K
// step => 128 v_wmma_f32_16x16x32_bf16 per wave. A fragments (x rows,
// L2-hot 2MB) are two 16B global loads per K step.
//
// CDNA5 16-bit fragment layouts (wave32):
//   A 16x32: lane l holds row m=l&15; half=l>>4 -> K chunks
//            {kk+8h..+7} (v0-3) and {kk+16+8h..+7} (v4-7).
//   B 32x16: lane l holds col n=l&15; half -> K chunks {kk+8h..+7} and {kk+16+8h..+7}.
// ---------------------------------------------------------------------------
__global__ __launch_bounds__(256)
void mm_gemm_argmm_kernel(const uint16_t* __restrict__ xb,
                          const uint16_t* __restrict__ fqb,
                          unsigned long long* __restrict__ pmax,
                          unsigned long long* __restrict__ pmin) {
    __shared__ uint16_t tileB[BLK_N * ROW_ELT]; // 130 KB (<=320KB/WGP on gfx1250)

    const int lane = threadIdx.x & 31;
    const int wave = threadIdx.x >> 5;

    const int nBlocksN = K_DIM / BLK_N;                 // 512
    const int rowBlk   = (int)(blockIdx.x / nBlocksN) * BLK_M;
    const int colBase  = (int)(blockIdx.x % nBlocksN) * BLK_N;

    // ---- stage feature_queue tile into LDS (async, coalesced) ----
    {
        const uint32_t ldsBase = (uint32_t)(uintptr_t)(&tileB[0]);
        const int chunksTotal = BLK_N * (F_DIM / 8);    // 8192 x 16B
#pragma unroll 4
        for (int c = threadIdx.x; c < chunksTotal; c += 256) {
            const int r = c >> 6;                       // 64 chunks per row
            const int j = c & 63;
            const uint32_t lds_off =
                ldsBase + (uint32_t)(r * ROW_ELT + j * 8) * 2u;
            const uint64_t gaddr =
                (uint64_t)(uintptr_t)(fqb + (size_t)(colBase + r) * F_DIM + j * 8);
            asm volatile("global_load_async_to_lds_b128 %0, %1, off"
                         :: "v"(lds_off), "v"(gaddr)
                         : "memory");
        }
        asm volatile("s_wait_asynccnt 0x0" ::: "memory");
    }
    __syncthreads();

    const int lm   = lane & 15;
    const int half = lane >> 4;
    const int rowBase = rowBlk + wave * 16;

    f32x8 acc[NT];
#pragma unroll
    for (int t = 0; t < NT; ++t)
#pragma unroll
        for (int r = 0; r < 8; ++r) acc[t][r] = 0.0f;

    const uint16_t* aRow = xb + (size_t)(rowBase + lm) * F_DIM;

    for (int kk = 0; kk < F_DIM; kk += KSTEP) {
        union { bf16x16 v; bf16x8 h[2]; } A;
        A.h[0] = *reinterpret_cast<const bf16x8*>(aRow + kk + half * 8);
        A.h[1] = *reinterpret_cast<const bf16x8*>(aRow + kk + 16 + half * 8);
#pragma unroll
        for (int t = 0; t < NT; ++t) {
            const uint16_t* bp =
                &tileB[(t * 16 + lm) * ROW_ELT + kk + half * 8];
            union { bf16x16 v; bf16x8 h[2]; } Bf;
            Bf.h[0] = *reinterpret_cast<const bf16x8*>(bp);
            Bf.h[1] = *reinterpret_cast<const bf16x8*>(bp + 16);
            acc[t] = __builtin_amdgcn_wmma_f32_16x16x32_bf16(
                false, A.v, false, Bf.v, (short)0, acc[t], false, false);
        }
    }

    // C/D layout: VGPR r, lane l -> row m = r + 8*(l>>4), col n = l&15.
#pragma unroll
    for (int r = 0; r < 8; ++r) {
        float maxv = acc[0][r]; int maxk = colBase + lm;
        float minv = maxv;      int mink = maxk;
#pragma unroll
        for (int t = 1; t < NT; ++t) {
            float v = acc[t][r];
            int   k = colBase + t * 16 + lm;
            if (v > maxv) { maxv = v; maxk = k; }
            if (v < minv) { minv = v; mink = k; }
        }
        // Cross-lane tree within each 16-lane group (wave32 shuffles).
#pragma unroll
        for (int m = 1; m < 16; m <<= 1) {
            float ov = __shfl_xor(maxv, m, 32);
            int   ok = __shfl_xor(maxk, m, 32);
            if (ov > maxv) { maxv = ov; maxk = ok; }
            float iv = __shfl_xor(minv, m, 32);
            int   ik = __shfl_xor(mink, m, 32);
            if (iv < minv) { minv = iv; mink = ik; }
        }
        if (lm == 0) {
            int b = rowBase + r + half * 8;
            unsigned long long pk =
                ((unsigned long long)f32_ordkey(maxv) << 32) | (uint32_t)maxk;
            unsigned long long nk =
                ((unsigned long long)f32_ordkey(minv) << 32) | (uint32_t)mink;
            atomicMax(&pmax[b], pk);
            atomicMin(&pmin[b], nk);
        }
    }
}

// ---------------------------------------------------------------------------
// Gather: out = [pos_rep (B,F) ; neg_rep (B,F)] as float4 row copies.
// ---------------------------------------------------------------------------
__global__ __launch_bounds__(256)
void mm_gather_kernel(const float* __restrict__ rep,
                      const unsigned long long* __restrict__ pmax,
                      const unsigned long long* __restrict__ pmin,
                      float* __restrict__ out) {
    const int b = blockIdx.x;
    const int t = threadIdx.x;
    const uint32_t pidx = (uint32_t)(pmax[b] & 0xFFFFFFFFull);
    const uint32_t nidx = (uint32_t)(pmin[b] & 0xFFFFFFFFull);
    const float4* prow = reinterpret_cast<const float4*>(rep + (size_t)pidx * F_DIM);
    const float4* nrow = reinterpret_cast<const float4*>(rep + (size_t)nidx * F_DIM);
    float4* o = reinterpret_cast<float4*>(out);
    const int QF4 = F_DIM / 4; // 128
    if (t < QF4) {
        o[(size_t)b * QF4 + t] = prow[t];
    } else {
        o[(size_t)(B_DIM + b) * QF4 + (t - QF4)] = nrow[t - QF4];
    }
}

// ---------------------------------------------------------------------------
// Workspace layout in d_ws:
//   [0, 16KB)             : packed argmax u64[2048]
//   [16KB, 32KB)          : packed argmin u64[2048]
//   [32KB, 32KB+2MB)      : x bf16 (2048*512)
//   [32KB+2MB, +64MB)     : feature_queue bf16 (65536*512)
// ---------------------------------------------------------------------------
extern "C" void kernel_launch(void* const* d_in, const int* in_sizes, int n_in,
                              void* d_out, int out_size, void* d_ws, size_t ws_size,
                              hipStream_t stream) {
    const float* x   = (const float*)d_in[0];
    const float* fq  = (const float*)d_in[1];
    const float* rep = (const float*)d_in[2];
    float* out = (float*)d_out;

    uint8_t* ws = (uint8_t*)d_ws;
    unsigned long long* pmax = (unsigned long long*)ws;
    unsigned long long* pmin = (unsigned long long*)(ws + (size_t)B_DIM * 8);
    uint16_t* xb  = (uint16_t*)(ws + (size_t)2 * B_DIM * 8);
    uint16_t* fqb = (uint16_t*)(ws + (size_t)2 * B_DIM * 8 +
                                (size_t)B_DIM * F_DIM * sizeof(uint16_t));

    // 1) init reduction slots
    mm_init_kernel<<<(B_DIM + 255) / 256, 256, 0, stream>>>(pmax, pmin);

    // 2) fp32 -> bf16 conversion passes
    {
        int n4 = (B_DIM * F_DIM) / 4;
        mm_cvt4_kernel<<<(n4 + 255) / 256, 256, 0, stream>>>(
            (const float4*)x, (ushort4*)xb, n4);
    }
    {
        int n4 = (K_DIM * F_DIM) / 4;
        mm_cvt4_kernel<<<(n4 + 255) / 256, 256, 0, stream>>>(
            (const float4*)fq, (ushort4*)fqb, n4);
    }

    // 3) fused WMMA GEMM + argmax/argmin (LDS-staged B tiles)
    {
        int blocks = (B_DIM / BLK_M) * (K_DIM / BLK_N); // 16 * 512 = 8192
        mm_gemm_argmm_kernel<<<blocks, 256, 0, stream>>>(xb, fqb, pmax, pmin);
    }

    // 4) gather pos/neg representation rows
    mm_gather_kernel<<<B_DIM, 256, 0, stream>>>(rep, pmax, pmin, out);
}
